// EncoderMeshBlock_10763188043860
// MI455X (gfx1250) — matrix operand order repaired and checked by reading, not compiled, not run on hardware
//
#include <hip/hip_runtime.h>

typedef __attribute__((ext_vector_type(16))) _Float16 v16h;
typedef __attribute__((ext_vector_type(8)))  _Float16 v8h;
typedef __attribute__((ext_vector_type(4)))  _Float16 v4h;
typedef __attribute__((ext_vector_type(8)))  float    v8f;

static constexpr int VN   = 100000;   // vertices
static constexpr int FN   = 200000;   // faces
static constexpr int PN   = 600000;   // incidence pairs = 3*F
static constexpr int KBIN = 27;       // fuzzy kernel bins
static constexpr int KPAD = 32;       // padded to WMMA K granularity
static constexpr float EPSV = 1e-5f;

// ---------------------------------------------------------------------------
// B operands are PRE-PACKED into the WMMA 32x16 f16 fragment layout:
//   fragment id = kt * NTILES + nt ; within a fragment each lane owns 16
//   contiguous halves (32 bytes) -> fragment load is one 32B access.
// Lane mapping (ISA 7.12.2): lanes 0-15 -> N=lane,  K 0..7 then 16..23
//                            lanes 16-31 -> N=lane-16, K 8..15 then 24..31
// ---------------------------------------------------------------------------
__global__ void pack_b_kernel(const float* __restrict__ src, _Float16* __restrict__ dst,
                              int Ksrc, int N, long total)
{
  long i = (long)blockIdx.x * blockDim.x + threadIdx.x;
  if (i >= total) return;                  // total = (Kpad/32) * (N/16) * 256
  const int j    = (int)(i & 7);
  const int lane = (int)((i >> 3) & 31);
  const int frag = (int)(i >> 8);
  const int ntiles = N >> 4;
  const int kt = frag / ntiles, nt = frag % ntiles;
  const int n  = nt * 16 + (lane & 15);
  const int k0 = kt * 32 + ((lane >> 4) << 3) + 2 * j + ((j >> 2) << 3);
  _Float16 a = (k0     < Ksrc) ? (_Float16)src[(long)k0       * N + n] : (_Float16)0.f;
  _Float16 b = (k0 + 1 < Ksrc) ? (_Float16)src[(long)(k0 + 1) * N + n] : (_Float16)0.f;
  _Float16* d = dst + (long)frag * 512 + lane * 16 + 2 * j;
  d[0] = a; d[1] = b;
}

// A fragment: row-major [M,K] f16, lane-contiguous -> two b128 loads.
__device__ inline v16h load_a_frag(const _Float16* __restrict__ aptr /* row base + lane K-offset */)
{
  v8h lo = *(const v8h*)(aptr);
  v8h hi = *(const v8h*)(aptr + 16);
  return __builtin_shufflevector(lo, hi, 0,1,2,3,4,5,6,7,8,9,10,11,12,13,14,15);
}

// ---------------- GEMM (f16 in, f32 out) + per-channel BN statistics ----------------
// B fragments staged in LDS (shared by all 8 waves); BN stats reduced in LDS
// first (ds_add_f32), then ONE global atomic per channel per block.
template<int NTILES>
__global__ __launch_bounds__(256)
void gemm_stats_kernel(const _Float16* __restrict__ A, const _Float16* __restrict__ Bp,
                       float* __restrict__ Y, float* __restrict__ stats, int M, int K)
{
  constexpr int N = NTILES * 16;
  constexpr int MAXKT = 4;                       // K <= 128
  __shared__ _Float16 Bs[MAXKT * NTILES * 512];  // packed fragments
  __shared__ float sstats[2 * N];                // [sum | sumsq]

  const int tid    = threadIdx.x;
  const int lane   = tid & 31;
  const int wave   = tid >> 5;
  const int tile_m = blockIdx.x * 8 + wave;
  const bool active = (tile_m * 16 < M);         // wave-uniform
  const int ktiles = K >> 5;

  // cooperative stage: packed B -> LDS (16B per thread per step), zero stats
  const int chunks = (ktiles * NTILES * 512) / 8;
  for (int idx = tid; idx < chunks; idx += 256)
    ((v8h*)Bs)[idx] = ((const v8h*)Bp)[idx];
  for (int idx = tid; idx < 2 * N; idx += 256)
    sstats[idx] = 0.f;
  __syncthreads();

  if (active) {
    const int arow = tile_m * 16 + (lane & 15);
    const _Float16* aptr = A + (long)arow * K + ((lane >> 4) << 3);

    v8f acc[NTILES];
    const v8f zero8 = {0.f,0.f,0.f,0.f,0.f,0.f,0.f,0.f};
#pragma unroll
    for (int nt = 0; nt < NTILES; ++nt) acc[nt] = zero8;

    for (int kt = 0; kt < ktiles; ++kt) {
      if (kt + 1 < ktiles)
        __builtin_prefetch(aptr + (kt + 1) * 32, 0, 1);   // global_prefetch_b8
      const v16h a = load_a_frag(aptr + kt * 32);
      const _Float16* bb = Bs + kt * NTILES * 512 + lane * 16;
#pragma unroll
      for (int nt = 0; nt < NTILES; ++nt) {
        const v16h b = *(const v16h*)(bb + nt * 512);      // ds_load_b128 x2
        acc[nt] = __builtin_amdgcn_wmma_f32_16x16x32_f16(false, a, false, b,
                                                         (short)0, acc[nt], false, false);
      }
    }

    // C/D layout: lanes 0-15 -> N=lane, M=VGPRidx ; lanes 16-31 -> N=lane-16, M=VGPRidx+8
    const int rowbase = tile_m * 16 + ((lane >> 4) << 3);
#pragma unroll
    for (int nt = 0; nt < NTILES; ++nt) {
      const int col = nt * 16 + (lane & 15);
      float lsum = 0.f, lsq = 0.f;
#pragma unroll
      for (int g = 0; g < 8; ++g) {
        const float v = acc[nt][g];
        Y[(long)(rowbase + g) * N + col] = v;
        lsum += v; lsq += v * v;
      }
      atomicAdd(&sstats[col],     lsum);   // LDS atomics (ds_add_f32)
      atomicAdd(&sstats[N + col], lsq);
    }
  }
  __syncthreads();
  for (int idx = tid; idx < 2 * N; idx += 256)             // 1 global atomic / channel / block
    atomicAdd(&stats[idx], sstats[idx]);
}

// ---------------- fused f2v: per-pair filter WMMA + gather(BN+ReLU facet) + scatter ----------------
__global__ __launch_bounds__(256)
void f2v_wmma_scatter_kernel(const _Float16* __restrict__ filt16, const _Float16* __restrict__ WkP,
                             const float* __restrict__ Yf, const float* __restrict__ scale,
                             const float* __restrict__ shift, const int* __restrict__ vt_map,
                             float* __restrict__ s, int Pn)
{
  constexpr int C = 128;
  __shared__ _Float16 Wks[8 * 512];    // packed Wk fragments (8 KB)
  const int tid  = threadIdx.x;
  const int lane = tid & 31;
  const int wave = tid >> 5;
  const int tile = blockIdx.x * 8 + wave;

  for (int idx = tid; idx < (8 * 512) / 8; idx += 256)
    ((v8h*)Wks)[idx] = ((const v8h*)WkP)[idx];
  __syncthreads();

  if (tile * 16 >= Pn) return;         // after the barrier: safe

  const int arow = tile * 16 + (lane & 15);
  const v16h a = load_a_frag(filt16 + (long)arow * KPAD + ((lane >> 4) << 3));

  const int rowbase = tile * 16 + ((lane >> 4) << 3);
  int vtx[8], fct[8];
#pragma unroll
  for (int g = 0; g < 8; ++g) {        // b64 pair loads
    int2 vf = *(const int2*)(vt_map + 2 * (rowbase + g));
    vtx[g] = vf.x; fct[g] = vf.y;
  }

#pragma unroll
  for (int nt = 0; nt < C / 16; ++nt) {
    const int col = nt * 16 + (lane & 15);
    const v16h b = *(const v16h*)(Wks + nt * 512 + lane * 16);
    v8f z = {0.f,0.f,0.f,0.f,0.f,0.f,0.f,0.f};
    const v8f w = __builtin_amdgcn_wmma_f32_16x16x32_f16(false, a, false, b, (short)0, z, false, false);
    const float sc = scale[col], sh = shift[col];
#pragma unroll
    for (int g = 0; g < 8; ++g) {
      const float xf = fmaxf(Yf[(long)fct[g] * C + col] * sc + sh, 0.f);
      atomicAdd(&s[(long)vtx[g] * C + col], xf * w[g]);
    }
  }
}

// ---------------- elementwise helpers (float4-wide) ----------------

__global__ void zero_f32v4_kernel(float4* __restrict__ p, long n4) {
  long i = (long)blockIdx.x * blockDim.x + threadIdx.x;
  if (i < n4) p[i] = make_float4(0.f, 0.f, 0.f, 0.f);
}

__global__ void copy_to_strided4_kernel(const float4* __restrict__ src, float* __restrict__ dst,
                                        int srcC4, int dstStride, long total4) {
  long i = (long)blockIdx.x * blockDim.x + threadIdx.x;
  if (i >= total4) return;
  long r = i / srcC4; int c4 = (int)(i % srcC4);
  *(float4*)(dst + r * dstStride + 4 * c4) = src[i];
}

// vertex->facet depthwise gather: dw[f,c] = sum_k x[face[f,k],c]*Wd[k,c]  (4 channels/thread)
__global__ void gather_dw4_kernel(const float* __restrict__ x, int xstride,
                                  const int* __restrict__ face, const float* __restrict__ Wd,
                                  _Float16* __restrict__ out, int ci, long total4) {
  long i = (long)blockIdx.x * blockDim.x + threadIdx.x;
  if (i >= total4) return;
  const int ci4 = ci >> 2;
  const int c4  = (int)(i % ci4);
  const long f  = i / ci4;
  float4 acc = make_float4(0.f, 0.f, 0.f, 0.f);
#pragma unroll
  for (int k = 0; k < 3; ++k) {
    const int vtx = face[f * 3 + k];
    const float4 xv = *(const float4*)(x + (long)vtx * xstride + 4 * c4);
    const float4 wd = *(const float4*)(Wd + k * ci + 4 * c4);
    acc.x += xv.x * wd.x; acc.y += xv.y * wd.y;
    acc.z += xv.z * wd.z; acc.w += xv.w * wd.w;
  }
  v4h o; o[0] = (_Float16)acc.x; o[1] = (_Float16)acc.y;
         o[2] = (_Float16)acc.z; o[3] = (_Float16)acc.w;
  *(v4h*)(out + f * ci + 4 * c4) = o;
}

__global__ void cvt_f16v4_kernel(const float4* __restrict__ src, _Float16* __restrict__ dst, long n4) {
  long i = (long)blockIdx.x * blockDim.x + threadIdx.x;
  if (i >= n4) return;
  const float4 v = src[i];
  v4h o; o[0] = (_Float16)v.x; o[1] = (_Float16)v.y; o[2] = (_Float16)v.z; o[3] = (_Float16)v.w;
  *(v4h*)(dst + 4 * i) = o;
}

// filt_coeff [P,27] -> f16 [P,32] zero-padded (4 bins/thread)
__global__ void pad_filt4_kernel(const float* __restrict__ filt, _Float16* __restrict__ dst, long total4) {
  long i = (long)blockIdx.x * blockDim.x + threadIdx.x;
  if (i >= total4) return;
  const long p = i >> 3;
  const int  k0 = (int)(i & 7) * 4;
  v4h o;
#pragma unroll
  for (int t = 0; t < 4; ++t)
    o[t] = (k0 + t < KBIN) ? (_Float16)filt[p * KBIN + k0 + t] : (_Float16)0.f;
  *(v4h*)(dst + p * KPAD + k0) = o;
}

// s[v,c] / max(nf[v],1) -> f16 A matrix (4 channels/thread, C = 128)
__global__ void div_nf_f16v4_kernel(const float* __restrict__ s, const int* __restrict__ nf,
                                    _Float16* __restrict__ dst, long total4) {
  long i = (long)blockIdx.x * blockDim.x + threadIdx.x;
  if (i >= total4) return;
  const long v = i >> 5;                 // 32 float4 per row of 128
  int n = nf[v]; if (n < 1) n = 1;
  const float inv = 1.f / (float)n;
  const float4 sv = ((const float4*)s)[i];
  v4h o; o[0] = (_Float16)(sv.x * inv); o[1] = (_Float16)(sv.y * inv);
         o[2] = (_Float16)(sv.z * inv); o[3] = (_Float16)(sv.w * inv);
  *(v4h*)(dst + 4 * i) = o;
}

__global__ void bn_finalize_kernel(const float* __restrict__ stats, const float* __restrict__ gamma,
                                   const float* __restrict__ beta, float* __restrict__ scale,
                                   float* __restrict__ shift, int N, float invM) {
  int n = blockIdx.x * blockDim.x + threadIdx.x;
  if (n >= N) return;
  float mean = stats[n] * invM;
  float var  = stats[N + n] * invM - mean * mean;
  float inv  = rsqrtf(var + EPSV);
  float sc   = gamma[n] * inv;
  scale[n] = sc;
  shift[n] = beta[n] - mean * sc;
}

__global__ void bn_relu_store4_kernel(const float4* __restrict__ Y, const float* __restrict__ scale,
                                      const float* __restrict__ shift, float* __restrict__ dst,
                                      int dstStride, int dstOff, int N, long total4) {
  long i = (long)blockIdx.x * blockDim.x + threadIdx.x;
  if (i >= total4) return;
  const int N4 = N >> 2;
  const long r = i / N4; const int c4 = (int)(i % N4);
  const float4 y  = Y[i];
  const float4 sc = *(const float4*)(scale + 4 * c4);
  const float4 sh = *(const float4*)(shift + 4 * c4);
  float4 o;
  o.x = fmaxf(y.x * sc.x + sh.x, 0.f);
  o.y = fmaxf(y.y * sc.y + sh.y, 0.f);
  o.z = fmaxf(y.z * sc.z + sh.z, 0.f);
  o.w = fmaxf(y.w * sc.w + sh.w, 0.f);
  *(float4*)(dst + r * dstStride + dstOff + 4 * c4) = o;
}

// ---------------- host sequencing ----------------

extern "C" void kernel_launch(void* const* d_in, const int* in_sizes, int n_in,
                              void* d_out, int out_size, void* d_ws, size_t ws_size,
                              hipStream_t stream)
{
  const float* inputs = (const float*)d_in[0];   // [V,64]
  const int*   face   = (const int*)  d_in[1];   // [F,3]
  const int*   nf     = (const int*)  d_in[2];   // [V]
  const int*   vt_map = (const int*)  d_in[3];   // [P,2]
  const float* filt   = (const float*)d_in[4];   // [P,27]

  struct ConvP { const float *Wd, *Wk, *Wp1, *Wp2, *b1, *b2, *g1, *g2; };
  ConvP cp[4];
  for (int k = 0; k < 4; ++k) {                  // sorted-key leaf order per conv dict
    int b = 5 + 8 * k;
    cp[k] = { (const float*)d_in[b+0], (const float*)d_in[b+1], (const float*)d_in[b+2],
              (const float*)d_in[b+3], (const float*)d_in[b+4], (const float*)d_in[b+5],
              (const float*)d_in[b+6], (const float*)d_in[b+7] };
  }
  const float* tW = (const float*)d_in[37];
  const float* tb = (const float*)d_in[38];
  const float* tg = (const float*)d_in[39];

  // workspace carve-out (256B aligned)
  char* ws = (char*)d_ws; size_t off = 0;
  auto take = [&](size_t bytes) -> void* {
    void* p = ws + off; off += (bytes + 255) & ~(size_t)255; return p;
  };
  float*    xbuf   = (float*)   take((size_t)VN * 128 * 4);   // DenseNet feature buffer
  float*    netbuf = (float*)   take((size_t)VN * 128 * 4);   // intra-iteration activations
  _Float16* dwA    = (_Float16*)take((size_t)FN * 128 * 2);   // v2f depthwise result (A matrix)
  float*    Yf     = (float*)   take((size_t)FN * 128 * 4);   // facet pre-BN GEMM output
  _Float16* filt16 = (_Float16*)take((size_t)PN * KPAD * 2);  // padded fuzzy coeffs (A matrix)
  float*    sbuf   = (float*)   take((size_t)VN * 128 * 4);   // vertex scatter accumulator
  _Float16* sA16   = (_Float16*)take((size_t)VN * 128 * 2);   // s/denom (A matrix), reused for transit
  float*    Yv     = (float*)   take((size_t)VN * 128 * 4);   // vertex pre-BN GEMM output
  _Float16* BpW    = (_Float16*)take((size_t)128 * 128 * 2);  // packed pointwise weights (fragments)
  _Float16* WkP    = (_Float16*)take((size_t)KPAD * 128 * 2); // packed padded Wk (fragments)
  float*    stats  = (float*)   take(256 * 4);                // [sum(128) | sumsq(128)]
  float*    scale  = (float*)   take(128 * 4);
  float*    shift  = (float*)   take(128 * 4);

  const int TB = 256;
  auto blocks = [](long n, int b) { return dim3((unsigned)((n + b - 1) / b)); };
  auto gemm_blocks = [](int M) { return dim3((unsigned)((M / 16 + 7) / 8)); };
  auto pack_total = [](int Kpad, int N) { return (long)(Kpad / 32) * (N / 16) * 256; };

  // x[:, :64] = inputs
  copy_to_strided4_kernel<<<blocks((long)VN * 16, TB), TB, 0, stream>>>(
      (const float4*)inputs, xbuf, 16, 128, (long)VN * 16);
  // filt -> f16 padded (reused by all 4 f2v stages)
  pad_filt4_kernel<<<blocks((long)PN * 8, TB), TB, 0, stream>>>(filt, filt16, (long)PN * 8);

  struct Stage { int pi; const float* in; int ci; float* outdst; int outOff; int co; };
  const Stage st[4] = {
    { 0, xbuf,   64,  netbuf, 0,  128 },   // conv0_0
    { 1, netbuf, 128, xbuf,   64, 32  },   // conv0_1 -> concat @ col 64
    { 2, xbuf,   96,  netbuf, 0,  128 },   // conv1_0 (reads x[:, :96])
    { 3, netbuf, 128, xbuf,   96, 32  },   // conv1_1 -> concat @ col 96
  };

  for (int sidx = 0; sidx < 4; ++sidx) {
    const Stage& S = st[sidx];
    const ConvP& P = cp[S.pi];
    const int ci = S.ci, mid = 128, co = S.co;

    // ---- v2f: gather+depthwise, then dw @ Wp1 with BN stats ----
    gather_dw4_kernel<<<blocks((long)FN * (ci / 4), TB), TB, 0, stream>>>(
        S.in, 128, face, P.Wd, dwA, ci, (long)FN * (ci / 4));
    pack_b_kernel<<<blocks(pack_total(ci, mid), TB), TB, 0, stream>>>(
        P.Wp1, BpW, ci, mid, pack_total(ci, mid));
    zero_f32v4_kernel<<<1, TB, 0, stream>>>((float4*)stats, 64);
    gemm_stats_kernel<8><<<gemm_blocks(FN), TB, 0, stream>>>(dwA, BpW, Yf, stats, FN, ci);
    bn_finalize_kernel<<<1, 128, 0, stream>>>(stats, P.g1, P.b1, scale, shift, mid, 1.0f / (float)FN);

    // ---- f2v: fused per-pair filter WMMA + BN/ReLU gather + atomic scatter ----
    pack_b_kernel<<<blocks(pack_total(KPAD, mid), TB), TB, 0, stream>>>(
        P.Wk, WkP, KBIN, mid, pack_total(KPAD, mid));
    zero_f32v4_kernel<<<blocks((long)VN * 32, TB), TB, 0, stream>>>((float4*)sbuf, (long)VN * 32);
    f2v_wmma_scatter_kernel<<<gemm_blocks(PN), TB, 0, stream>>>(
        filt16, WkP, Yf, scale, shift, vt_map, sbuf, PN);

    // ---- mean over incident faces, then s @ Wp2 with BN stats ----
    div_nf_f16v4_kernel<<<blocks((long)VN * 32, TB), TB, 0, stream>>>(sbuf, nf, sA16, (long)VN * 32);
    pack_b_kernel<<<blocks(pack_total(mid, co), TB), TB, 0, stream>>>(
        P.Wp2, BpW, mid, co, pack_total(mid, co));
    zero_f32v4_kernel<<<1, TB, 0, stream>>>((float4*)stats, 64);
    if (co == 128)
      gemm_stats_kernel<8><<<gemm_blocks(VN), TB, 0, stream>>>(sA16, BpW, Yv, stats, VN, mid);
    else
      gemm_stats_kernel<2><<<gemm_blocks(VN), TB, 0, stream>>>(sA16, BpW, Yv, stats, VN, mid);
    bn_finalize_kernel<<<1, 128, 0, stream>>>(stats, P.g2, P.b2, scale, shift, co, 1.0f / (float)VN);
    bn_relu_store4_kernel<<<blocks((long)VN * (co / 4), TB), TB, 0, stream>>>(
        (const float4*)Yv, scale, shift, S.outdst, 128, S.outOff, co, (long)VN * (co / 4));
  }

  // ---- transit: x @ W, BN, ReLU -> d_out ----
  cvt_f16v4_kernel<<<blocks((long)VN * 32, TB), TB, 0, stream>>>((const float4*)xbuf, sA16, (long)VN * 32);
  pack_b_kernel<<<blocks(pack_total(128, 128), TB), TB, 0, stream>>>(
      tW, BpW, 128, 128, pack_total(128, 128));
  zero_f32v4_kernel<<<1, TB, 0, stream>>>((float4*)stats, 64);
  gemm_stats_kernel<8><<<gemm_blocks(VN), TB, 0, stream>>>(sA16, BpW, Yv, stats, VN, 128);
  bn_finalize_kernel<<<1, 128, 0, stream>>>(stats, tg, tb, scale, shift, 128, 1.0f / (float)VN);
  bn_relu_store4_kernel<<<blocks((long)VN * 32, TB), TB, 0, stream>>>(
      (const float4*)Yv, scale, shift, (float*)d_out, 128, 0, 128, (long)VN * 32);
}
